// EncoderDecoder_16071767621856
// MI455X (gfx1250) — matrix operand order repaired and checked by reading, not compiled, not run on hardware
//
#include <hip/hip_runtime.h>
#include <hip/hip_bf16.h>
#include <math.h>

typedef __attribute__((ext_vector_type(16))) _Float16 v16h;
typedef __attribute__((ext_vector_type(8)))  float    v8f;

#define HH 512
#define BB 64
#define TSS 64
#define TTT 51
#define VTT 32000

__device__ __forceinline__ float sigm_(float x) { return 1.0f / (1.0f + expf(-x)); }

// ---------------------------------------------------------------------------
// Blocked f16 WMMA GEMM: one wave computes a 64(M) x 16(N) strip using 4
// accumulators; the B fragment is loaded once per K-step and reused by 4
// WMMAs (4x less traffic on the dominant weight stream, which stays
// L2-resident on the 192MB L2).  A: (M x K) row-major f16, M % 64 == 0.
// B: (N x K) row-major f16 (weight (out,in)), so the WMMA B operand (K x N)
// is B^T and fragment loads are contiguous b128s.
// Fragment layouts per CDNA5 ISA 7.12.2 (16-bit A 16x32, B 32x16, f32 C/D).
// ---------------------------------------------------------------------------
__device__ __forceinline__ void wmma_strip(
    const _Float16* __restrict__ A, const _Float16* __restrict__ B, int K,
    int mrow0, int nrow, int hi, v8f* acc)
{
    const _Float16* Brow = B + (size_t)nrow * K;
    for (int k0 = 0; k0 < K; k0 += 32) {
        v16h b;
        const int kb = k0 + hi * 8;    // B: lanes0-15 K=0..7,16..23, lanes16-31 K=8..15,24..31
        #pragma unroll
        for (int j = 0; j < 8; ++j)  b[j] = Brow[kb + j];
        #pragma unroll
        for (int j = 8; j < 16; ++j) b[j] = Brow[kb + 8 + j];
        const int ka = k0 + hi * 8;    // A: lanes0-15 K base 0, lanes16-31 base 8
        #pragma unroll
        for (int mi = 0; mi < 4; ++mi) {
            const _Float16* Arow = A + (size_t)(mrow0 + mi * 16) * K;
            v16h a;
            #pragma unroll
            for (int j = 0; j < 8; ++j)  a[j] = Arow[ka + j];          // K=ka..ka+7
            #pragma unroll
            for (int j = 8; j < 16; ++j) a[j] = Arow[ka + 8 + j];      // K=ka+16..+23
            acc[mi] = __builtin_amdgcn_wmma_f32_16x16x32_f16(
                false, a, false, b, (short)0, acc[mi], false, false);
        }
    }
}

__global__ __launch_bounds__(32) void wmma_gemm(
    const _Float16* __restrict__ A1, const _Float16* __restrict__ B1, int K1,
    const _Float16* __restrict__ A2, const _Float16* __restrict__ B2, int K2,
    const float* __restrict__ bias1, const float* __restrict__ bias2,
    float* __restrict__ C, int ldc, int act)
{
    const int lane = threadIdx.x;          // 0..31
    const int nt = blockIdx.x;             // N tile (16 cols)
    const int mb = blockIdx.y;             // M block (64 rows)
    const int l15 = lane & 15;
    const int hi  = lane >> 4;
    v8f acc[4] = {};

    const int mrow0 = mb * 64 + l15;
    const int nrow  = nt * 16 + l15;
    wmma_strip(A1, B1, K1, mrow0, nrow, hi, acc);
    if (A2) wmma_strip(A2, B2, K2, mrow0, nrow, hi, acc);

    // C/D layout: VGPR r, lanes0-15 -> row r col lane; lanes16-31 -> row 8+r
    const int n = nt * 16 + l15;
    float bv = 0.0f;
    if (bias1) bv += bias1[n];
    if (bias2) bv += bias2[n];
    #pragma unroll
    for (int mi = 0; mi < 4; ++mi) {
        const int mbase = mb * 64 + mi * 16 + hi * 8;
        #pragma unroll
        for (int r = 0; r < 8; ++r) {
            float v = acc[mi][r] + bv;
            if (act == 1) v = tanhf(v);
            C[(size_t)(mbase + r) * ldc + n] = v;
        }
    }
}

// ---------------------------------------------------------------------------
// Elementwise helpers
// ---------------------------------------------------------------------------
__global__ void zero_f32(float* p, int n) {
    int i = blockIdx.x * blockDim.x + threadIdx.x;
    if (i < n) p[i] = 0.0f;
}

__global__ void cvt_f32_f16(const float* __restrict__ s, _Float16* __restrict__ d, int n) {
    int i = blockIdx.x * blockDim.x + threadIdx.x;
    if (i < n) d[i] = (_Float16)s[i];
}

// x[t][b][h] = (tok==PAD) ? 0 : emb[tok][h]   (PAD row forced to zero)
__global__ void embed_f16(const int* __restrict__ tok, const float* __restrict__ emb,
                          _Float16* __restrict__ out, int strideB, int nSteps) {
    int i = blockIdx.x * blockDim.x + threadIdx.x;
    int total = nSteps * BB * HH;
    if (i >= total) return;
    int t = i / (BB * HH);
    int r = i - t * (BB * HH);
    int b = r >> 9;
    int h = r & (HH - 1);
    int w = tok[b * strideB + t];
    out[i] = (w == 0) ? (_Float16)0.0f : (_Float16)emb[(size_t)w * HH + h];
}

// LSTM gate nonlinearity: g = [i f g o] chunks of H, per batch row of 4H.
__global__ void lstm_gates(const float* __restrict__ G,
                           float* __restrict__ h32, _Float16* __restrict__ h16,
                           float* __restrict__ c, float* __restrict__ hs) {
    int i = blockIdx.x * blockDim.x + threadIdx.x;
    if (i >= BB * HH) return;
    int b = i >> 9, j = i & (HH - 1);
    const float* g = G + (size_t)b * (4 * HH);
    float ig = sigm_(g[j]);
    float fg = sigm_(g[HH + j]);
    float gg = tanhf(g[2 * HH + j]);
    float og = sigm_(g[3 * HH + j]);
    float cn = fg * c[i] + ig * gg;
    float hn = og * tanhf(cn);
    c[i] = cn; h32[i] = hn; h16[i] = (_Float16)hn;
    if (hs) hs[i] = hn;
}

// enc_out = hs_f + hs_b (f32 + f16), hx/cx init = hT_f+hT_b / cT_f+cT_b
__global__ void enc_combine(const float* __restrict__ hsF, const float* __restrict__ hsB,
                            float* __restrict__ eo32, _Float16* __restrict__ eo16,
                            const float* __restrict__ hF, const float* __restrict__ hB,
                            const float* __restrict__ cF, const float* __restrict__ cB,
                            float* __restrict__ hx32, _Float16* __restrict__ hx16,
                            float* __restrict__ cx) {
    int i = blockIdx.x * blockDim.x + threadIdx.x;
    if (i < TSS * BB * HH) {
        float v = hsF[i] + hsB[i];
        eo32[i] = v; eo16[i] = (_Float16)v;
    }
    if (i < BB * HH) {
        float hv = hF[i] + hB[i];
        hx32[i] = hv; hx16[i] = (_Float16)hv;
        cx[i] = cF[i] + cB[i];
    }
}

// scores + softmax + (reference's transposed) mask.  block x = batch row,
// thread y = time.  attn[x][y] = softmax_y(score[x][.])[y] * (source[y][x]==PAD)
__global__ __launch_bounds__(64) void attn_scores(
    const float* __restrict__ encFeat /*[t][b][H]*/, const float* __restrict__ decF,
    const float* __restrict__ vw, const float* __restrict__ vb,
    const int* __restrict__ source, float* __restrict__ attn) {
    int x = blockIdx.x, y = threadIdx.x;
    const float* ef = encFeat + ((size_t)y * BB + x) * HH;
    const float* df = decF + (size_t)x * HH;
    float s = vb[0];
    for (int h = 0; h < HH; ++h) s += tanhf(ef[h] + df[h]) * vw[h];
    __shared__ float sm[64], se[64];
    sm[y] = s; __syncthreads();
    float m = -3.4e38f;
    for (int i = 0; i < 64; ++i) m = fmaxf(m, sm[i]);
    float e = expf(s - m);
    se[y] = e; __syncthreads();
    float sum = 0.0f;
    for (int i = 0; i < 64; ++i) sum += se[i];
    float mask = (source[y * TSS + x] == 0) ? 1.0f : 0.0f;  // reference reshape quirk
    attn[x * TSS + y] = (e / sum) * mask;
}

// content[y][h] = sum_x attn[x][y] * enc_out[x][y][h]; write cat=[content,hx] f16
__global__ __launch_bounds__(256) void attn_content(
    const float* __restrict__ attn, const float* __restrict__ eo32 /*[t][b][H]*/,
    const float* __restrict__ hx32, _Float16* __restrict__ cat) {
    int y = blockIdx.x;
    for (int h = threadIdx.x; h < HH; h += 256) {
        float s = 0.0f;
        for (int x = 0; x < BB; ++x)
            s += attn[x * TSS + y] * eo32[((size_t)y * BB + x) * HH + h];
        cat[(size_t)y * (2 * HH) + h] = (_Float16)s;
        cat[(size_t)y * (2 * HH) + HH + h] = (_Float16)hx32[(size_t)y * HH + h];
    }
}

// per-batch CE over 32000 logits: ceVal[b] = valid * (m + log(sumexp) - logit[wt])
__global__ __launch_bounds__(256) void ce_rows(
    const float* __restrict__ logits, const int* __restrict__ target, int t,
    float* __restrict__ ceVal, float* __restrict__ validVal) {
    int b = blockIdx.x;
    const float* L = logits + (size_t)b * VTT;
    __shared__ float red[256];
    float m = -3.4e38f;
    for (int i = threadIdx.x; i < VTT; i += 256) m = fmaxf(m, L[i]);
    red[threadIdx.x] = m; __syncthreads();
    for (int s = 128; s > 0; s >>= 1) {
        if (threadIdx.x < s) red[threadIdx.x] = fmaxf(red[threadIdx.x], red[threadIdx.x + s]);
        __syncthreads();
    }
    m = red[0]; __syncthreads();
    float acc = 0.0f;
    for (int i = threadIdx.x; i < VTT; i += 256) acc += expf(L[i] - m);
    red[threadIdx.x] = acc; __syncthreads();
    for (int s = 128; s > 0; s >>= 1) {
        if (threadIdx.x < s) red[threadIdx.x] += red[threadIdx.x + s];
        __syncthreads();
    }
    if (threadIdx.x == 0) {
        float sum = red[0];
        int wt = target[b * TTT + t + 1];
        float valid = (wt != 0) ? 1.0f : 0.0f;
        float lp = L[wt] - m - logf(sum);
        ceVal[b] = -lp * valid;
        validVal[b] = valid;
    }
}

__global__ __launch_bounds__(64) void loss_step(
    const float* __restrict__ ceVal, const float* __restrict__ validVal, float* out) {
    __shared__ float a[64], v[64];
    int i = threadIdx.x;
    a[i] = ceVal[i]; v[i] = validVal[i]; __syncthreads();
    if (i == 0) {
        float sa = 0.0f, sv = 0.0f;
        for (int k = 0; k < 64; ++k) { sa += a[k]; sv += v[k]; }
        float sl = (sv > 0.0f) ? (sa / fmaxf(sv, 1.0f)) : 0.0f;
        out[0] += sl;
    }
}

// ---------------------------------------------------------------------------
extern "C" void kernel_launch(void* const* d_in, const int* in_sizes, int n_in,
                              void* d_out, int out_size, void* d_ws, size_t ws_size,
                              hipStream_t stream) {
    (void)in_sizes; (void)n_in; (void)out_size; (void)ws_size;
    const int*   source    = (const int*)d_in[0];
    const int*   target    = (const int*)d_in[1];
    const float* emb_src   = (const float*)d_in[2];
    const float* emb_tgt   = (const float*)d_in[3];
    const float* encWihF   = (const float*)d_in[4];
    const float* encWhhF   = (const float*)d_in[5];
    const float* encBihF   = (const float*)d_in[6];
    const float* encBhhF   = (const float*)d_in[7];
    const float* encWihB   = (const float*)d_in[8];
    const float* encWhhB   = (const float*)d_in[9];
    const float* encBihB   = (const float*)d_in[10];
    const float* encBhhB   = (const float*)d_in[11];
    const float* Wh_w      = (const float*)d_in[12];
    const float* Wh_b      = (const float*)d_in[13];
    const float* decWih    = (const float*)d_in[14];
    const float* decWhh    = (const float*)d_in[15];
    const float* decBih    = (const float*)d_in[16];
    const float* decBhh    = (const float*)d_in[17];
    const float* Ws_w      = (const float*)d_in[18];
    const float* Ws_b      = (const float*)d_in[19];
    const float* v_w       = (const float*)d_in[20];
    const float* v_b       = (const float*)d_in[21];
    const float* lin_w     = (const float*)d_in[22];
    const float* lin_b     = (const float*)d_in[23];
    const float* out_w     = (const float*)d_in[24];
    const float* out_b     = (const float*)d_in[25];
    float* lossOut = (float*)d_out;

    char* base = (char*)d_ws; size_t off = 0;
    auto alloc = [&](size_t bytes) -> void* {
        void* r = base + off; off = (off + bytes + 255) & ~(size_t)255; return r;
    };
    // f16 weights (converted once per call; L2-resident thereafter)
    _Float16* wWihF   = (_Float16*)alloc((size_t)4*HH*HH*2);
    _Float16* wWhhF   = (_Float16*)alloc((size_t)4*HH*HH*2);
    _Float16* wWihB   = (_Float16*)alloc((size_t)4*HH*HH*2);
    _Float16* wWhhB   = (_Float16*)alloc((size_t)4*HH*HH*2);
    _Float16* wDecWih = (_Float16*)alloc((size_t)4*HH*HH*2);
    _Float16* wDecWhh = (_Float16*)alloc((size_t)4*HH*HH*2);
    _Float16* wWh     = (_Float16*)alloc((size_t)HH*HH*2);
    _Float16* wWs     = (_Float16*)alloc((size_t)HH*HH*2);
    _Float16* wLin    = (_Float16*)alloc((size_t)HH*2*HH*2);
    _Float16* wOut    = (_Float16*)alloc((size_t)VTT*HH*2);
    // embedded inputs [t][b][H] f16
    _Float16* xSrc    = (_Float16*)alloc((size_t)TSS*BB*HH*2);
    _Float16* xTgt    = (_Float16*)alloc((size_t)(TTT-1)*BB*HH*2);
    // encoder activations
    float* hsF        = (float*)alloc((size_t)TSS*BB*HH*4);
    float* hsB        = (float*)alloc((size_t)TSS*BB*HH*4);
    float* eo32       = (float*)alloc((size_t)TSS*BB*HH*4);
    _Float16* eo16    = (_Float16*)alloc((size_t)TSS*BB*HH*2);
    float* encFeat    = (float*)alloc((size_t)TSS*BB*HH*4);
    // recurrent states
    float*    hF32 = (float*)alloc(BB*HH*4);  _Float16* hF16 = (_Float16*)alloc(BB*HH*2);
    float*    cF   = (float*)alloc(BB*HH*4);
    float*    hB32 = (float*)alloc(BB*HH*4);  _Float16* hB16 = (_Float16*)alloc(BB*HH*2);
    float*    cB   = (float*)alloc(BB*HH*4);
    float*    hx32 = (float*)alloc(BB*HH*4);  _Float16* hx16 = (_Float16*)alloc(BB*HH*2);
    float*    cx   = (float*)alloc(BB*HH*4);
    // per-step scratch
    float* G       = (float*)alloc((size_t)BB*4*HH*4);
    float* decF    = (float*)alloc(BB*HH*4);
    float* attn    = (float*)alloc(BB*TSS*4);
    _Float16* cat  = (_Float16*)alloc((size_t)BB*2*HH*2);
    float* hxn32   = (float*)alloc(BB*HH*4);
    _Float16* hxn16= (_Float16*)alloc(BB*HH*2);
    float* logits  = (float*)alloc((size_t)BB*VTT*4);
    float* ceVal   = (float*)alloc(BB*4);
    float* valVal  = (float*)alloc(BB*4);

    const dim3 blk256(256);
    auto g1 = [](int n) { return dim3((n + 255) / 256); };

    // --- init ---
    zero_f32<<<1, 1, 0, stream>>>(lossOut, 1);
    zero_f32<<<g1(BB*HH), blk256, 0, stream>>>(hF32, BB*HH);
    zero_f32<<<g1(BB*HH), blk256, 0, stream>>>(cF, BB*HH);
    zero_f32<<<g1(BB*HH), blk256, 0, stream>>>(hB32, BB*HH);
    zero_f32<<<g1(BB*HH), blk256, 0, stream>>>(cB, BB*HH);
    zero_f32<<<g1(BB*HH/2), blk256, 0, stream>>>((float*)hF16, BB*HH/2);
    zero_f32<<<g1(BB*HH/2), blk256, 0, stream>>>((float*)hB16, BB*HH/2);

    // --- weight conversion fp32 -> f16 ---
    cvt_f32_f16<<<g1(4*HH*HH), blk256, 0, stream>>>(encWihF, wWihF, 4*HH*HH);
    cvt_f32_f16<<<g1(4*HH*HH), blk256, 0, stream>>>(encWhhF, wWhhF, 4*HH*HH);
    cvt_f32_f16<<<g1(4*HH*HH), blk256, 0, stream>>>(encWihB, wWihB, 4*HH*HH);
    cvt_f32_f16<<<g1(4*HH*HH), blk256, 0, stream>>>(encWhhB, wWhhB, 4*HH*HH);
    cvt_f32_f16<<<g1(4*HH*HH), blk256, 0, stream>>>(decWih, wDecWih, 4*HH*HH);
    cvt_f32_f16<<<g1(4*HH*HH), blk256, 0, stream>>>(decWhh, wDecWhh, 4*HH*HH);
    cvt_f32_f16<<<g1(HH*HH), blk256, 0, stream>>>(Wh_w, wWh, HH*HH);
    cvt_f32_f16<<<g1(HH*HH), blk256, 0, stream>>>(Ws_w, wWs, HH*HH);
    cvt_f32_f16<<<g1(2*HH*HH), blk256, 0, stream>>>(lin_w, wLin, 2*HH*HH);
    cvt_f32_f16<<<g1(VTT*HH), blk256, 0, stream>>>(out_w, wOut, VTT*HH);

    // --- embeddings ---
    embed_f16<<<g1(TSS*BB*HH), blk256, 0, stream>>>(source, emb_src, xSrc, TSS, TSS);
    embed_f16<<<g1((TTT-1)*BB*HH), blk256, 0, stream>>>(target, emb_tgt, xTgt, TTT, TTT-1);

    const dim3 gGate(4*HH/16, 1);         // N=2048 strip tiles, M=64 in one block row
    const dim3 gH(HH/16, 1);              // N=512
    const dim3 wblk(32);

    // --- encoder forward direction ---
    for (int t = 0; t < TSS; ++t) {
        wmma_gemm<<<gGate, wblk, 0, stream>>>(
            xSrc + (size_t)t*BB*HH, wWihF, HH, hF16, wWhhF, HH,
            encBihF, encBhhF, G, 4*HH, 0);
        lstm_gates<<<g1(BB*HH), blk256, 0, stream>>>(G, hF32, hF16, cF,
            hsF + (size_t)t*BB*HH);
    }
    // --- encoder backward direction (consumes x reversed; hs stored at true t) ---
    for (int s = 0; s < TSS; ++s) {
        int t = TSS - 1 - s;
        wmma_gemm<<<gGate, wblk, 0, stream>>>(
            xSrc + (size_t)t*BB*HH, wWihB, HH, hB16, wWhhB, HH,
            encBihB, encBhhB, G, 4*HH, 0);
        lstm_gates<<<g1(BB*HH), blk256, 0, stream>>>(G, hB32, hB16, cB,
            hsB + (size_t)t*BB*HH);
    }
    // --- combine directions; init decoder state ---
    enc_combine<<<g1(TSS*BB*HH), blk256, 0, stream>>>(
        hsF, hsB, eo32, eo16, hF32, hB32, cF, cB, hx32, hx16, cx);

    // --- enc_feat = enc_out @ Wh^T + Wh_b  (4096 x 512, K=512) ---
    {
        dim3 gF(HH/16, TSS*BB/64);
        wmma_gemm<<<gF, wblk, 0, stream>>>(
            eo16, wWh, HH, (const _Float16*)nullptr, (const _Float16*)nullptr, 0,
            Wh_b, (const float*)nullptr, encFeat, HH, 0);
    }

    // --- decoder loop ---
    const dim3 gLogit(VTT/16, 1);         // 2000 strip tiles
    for (int t = 0; t < TTT - 1; ++t) {
        // LSTM cell: g = e@Wih^T + hx@Whh^T + biases
        wmma_gemm<<<gGate, wblk, 0, stream>>>(
            xTgt + (size_t)t*BB*HH, wDecWih, HH, hx16, wDecWhh, HH,
            decBih, decBhh, G, 4*HH, 0);
        lstm_gates<<<g1(BB*HH), blk256, 0, stream>>>(G, hx32, hx16, cx, (float*)nullptr);
        // dec_f = hx @ Ws^T + Ws_b
        wmma_gemm<<<gH, wblk, 0, stream>>>(
            hx16, wWs, HH, (const _Float16*)nullptr, (const _Float16*)nullptr, 0,
            Ws_b, (const float*)nullptr, decF, HH, 0);
        // attention (replicates reference reshape semantics exactly)
        attn_scores<<<dim3(BB), dim3(64), 0, stream>>>(encFeat, decF, v_w, v_b, source, attn);
        attn_content<<<dim3(BB), blk256, 0, stream>>>(attn, eo32, hx32, cat);
        // hx_new = tanh(cat @ lin^T + lin_b)
        wmma_gemm<<<gH, wblk, 0, stream>>>(
            cat, wLin, 2*HH, (const _Float16*)nullptr, (const _Float16*)nullptr, 0,
            lin_b, (const float*)nullptr, hxn32, HH, 1);
        cvt_f32_f16<<<g1(BB*HH), blk256, 0, stream>>>(hxn32, hxn16, BB*HH);
        // logits = hx_new @ out_w^T + out_b   (dominant GEMM: 64x32000, K=512)
        wmma_gemm<<<gLogit, wblk, 0, stream>>>(
            hxn16, wOut, HH, (const _Float16*)nullptr, (const _Float16*)nullptr, 0,
            out_b, (const float*)nullptr, logits, VTT, 0);
        // masked mean CE for this step, accumulated into scalar loss
        ce_rows<<<dim3(BB), blk256, 0, stream>>>(logits, target, t, ceVal, valVal);
        loss_step<<<1, dim3(64), 0, stream>>>(ceVal, valVal, lossOut);
    }
}
